// DepthMarkerPredictor_71287867179113
// MI455X (gfx1250) — matrix operand order repaired and checked by reading, not compiled, not run on hardware
//
#include <hip/hip_runtime.h>
#include <hip/hip_bf16.h>

typedef __attribute__((ext_vector_type(16))) _Float16 v16h;
typedef __attribute__((ext_vector_type(8)))  _Float16 v8h;
typedef __attribute__((ext_vector_type(8)))  float    v8f;

#define HID      256
#define SEQ      512
#define BATCH    8192
#define KTILES   9          // K padded to 288: [h(256) | inp | 1 | 0...]
#define NTILES   65         // 64 gate tiles (i,f,g,o x 16) + 1 fc tile
#define WAVES    2          // waves per block; each wave owns 16 batch rows
#define HSTRIDE  296        // f16 per h row (288 + pad; 148 dwords -> conflict-free A loads)
#define CSTRIDE  260        // f32 per c row (256 + pad; 8*260 mod 64 == 32)

// ---------------------------------------------------------------------------
// Prologue: pack W_hh^T (+W_ih, +bias) and W_fc (+b_fc) into WMMA B-fragment
// layout. Fragment (nt,kt): 32 lanes x 16 f16, lane holds column n = nt*16 +
// (lane&15), K = kt*32 + (lane>>4)*8 + (e&7) + (e>>3)*16  (dense 16x16x32 f16 B layout).
// ---------------------------------------------------------------------------
__global__ void lstm_pack_weights(const float* __restrict__ W_ih,
                                  const float* __restrict__ W_hh,
                                  const float* __restrict__ b_ih,
                                  const float* __restrict__ b_hh,
                                  const float* __restrict__ W_fc,
                                  const float* __restrict__ b_fc,
                                  _Float16* __restrict__ wpack) {
  int tid = blockIdx.x * blockDim.x + threadIdx.x;
  if (tid >= NTILES * KTILES * 512) return;
  int fid = tid >> 9;            // fragment id
  int within = tid & 511;
  int l = within >> 4;           // lane 0..31
  int e = within & 15;           // element 0..15
  int nt = fid / KTILES;
  int kt = fid % KTILES;
  int k  = kt * 32 + (l >> 4) * 8 + (e & 7) + (e >> 3) * 16;
  float val = 0.0f;
  if (nt < 64) {
    int n = nt * 16 + (l & 15);  // gate row index in [0,1024)
    if (k < HID)        val = W_hh[n * HID + k];
    else if (k == HID)      val = W_ih[n];
    else if (k == HID + 1)  val = b_ih[n] + b_hh[n];
  } else {
    // fc tile: only column 0 is live
    if ((l & 15) == 0) {
      if (k < HID)            val = W_fc[k];
      else if (k == HID + 1)  val = b_fc[0];   // A has 1.0 at k=257 -> +b_fc
      // k == 256 (inp slot) stays 0: d does not depend on inp
    }
  }
  wpack[tid] = (_Float16)val;
}

// ---------------------------------------------------------------------------
// Main persistent-recurrence kernel.
// ---------------------------------------------------------------------------
__device__ __forceinline__ float fast_sig(float x) {
  return __builtin_amdgcn_rcpf(1.0f + __expf(-x));
}
__device__ __forceinline__ float fast_tanh(float x) {
  return 1.0f - 2.0f * __builtin_amdgcn_rcpf(__expf(2.0f * x) + 1.0f);
}

__global__ __launch_bounds__(32 * WAVES)
void DepthMarkerPredictor_lstm_kernel(const float* __restrict__ x,
                                      const _Float16* __restrict__ wpack,
                                      float* __restrict__ out) {
  __shared__ alignas(16) _Float16 hbuf[WAVES][16][HSTRIDE];
  __shared__ float                cbuf[WAVES][16][CSTRIDE];

  const int lane = threadIdx.x & 31;
  const int w    = threadIdx.x >> 5;
  const int sel  = lane >> 4;      // 0: lanes 0-15, 1: lanes 16-31
  const int nlo  = lane & 15;
  const int row_base = (blockIdx.x * WAVES + w) * 16;

  // ---- init state: h=0, c=0, A-col256 = x[row], A-col257 = 1 --------------
  _Float16* hflat = &hbuf[w][0][0];
  for (int i = lane; i < 16 * HSTRIDE; i += 32) hflat[i] = (_Float16)0.0f;
  float* cflat = &cbuf[w][0][0];
  for (int i = lane; i < 16 * CSTRIDE; i += 32) cflat[i] = 0.0f;
  __syncthreads();
  if (lane < 16) {
    hbuf[w][lane][HID]     = (_Float16)x[row_base + lane];
    hbuf[w][lane][HID + 1] = (_Float16)1.0f;
  }
  __syncthreads();

  // A-fragment loader: 16x32 f16 A layout — lane m=(lane&15); sel=0 holds
  // K [0..8)+[16..24), sel=1 holds K [8..16)+[24..32) of this k-tile.
  const _Float16* myrow = &hbuf[w][nlo][0];
#define LOAD_A(kt_)                                                        \
  ({ union { v16h v; v8h h2[2]; } u_;                                      \
     u_.h2[0] = *(const v8h*)(myrow + (kt_) * 32 + sel * 8);               \
     u_.h2[1] = *(const v8h*)(myrow + (kt_) * 32 + 16 + sel * 8);          \
     u_.v; })

  v16h a[KTILES];
#pragma unroll
  for (int kt = 0; kt < KTILES; ++kt) a[kt] = LOAD_A(kt);

  const int fragsz = 512;                       // f16 per fragment
  const _Float16* bbase = wpack + (lane << 4);  // lane-local 32B chunk

#pragma unroll 1
  for (int t = 0; t < SEQ; ++t) {
    // ---- gates: i/f/g/o tiles for each 16-unit hidden block --------------
#pragma unroll 1
    for (int jt = 0; jt < 16; ++jt) {
      v8f ai = {}, af = {}, ag = {}, ao = {};
#pragma unroll
      for (int kt = 0; kt < KTILES; ++kt) {
        const _Float16* bp = bbase + (jt * KTILES + kt) * fragsz;
        v16h bi = *(const v16h*)(bp);
        v16h bf = *(const v16h*)(bp + 16 * KTILES * fragsz);
        v16h bg = *(const v16h*)(bp + 32 * KTILES * fragsz);
        v16h bo = *(const v16h*)(bp + 48 * KTILES * fragsz);
        ai = __builtin_amdgcn_wmma_f32_16x16x32_f16(false, a[kt], false, bi,
                                                    (short)0, ai, false, false);
        af = __builtin_amdgcn_wmma_f32_16x16x32_f16(false, a[kt], false, bf,
                                                    (short)0, af, false, false);
        ag = __builtin_amdgcn_wmma_f32_16x16x32_f16(false, a[kt], false, bg,
                                                    (short)0, ag, false, false);
        ao = __builtin_amdgcn_wmma_f32_16x16x32_f16(false, a[kt], false, bo,
                                                    (short)0, ao, false, false);
      }
      // D layout: VGPR v -> row m = v + 8*sel, col n = jt*16 + nlo
      const int col = jt * 16 + nlo;
#pragma unroll
      for (int v = 0; v < 8; ++v) {
        const int m = v + 8 * sel;
        float co = cbuf[w][m][col];
        float cn = fast_sig(af[v]) * co + fast_sig(ai[v]) * fast_tanh(ag[v]);
        float hn = fast_sig(ao[v]) * fast_tanh(cn);
        cbuf[w][m][col] = cn;
        hbuf[w][m][col] = (_Float16)hn;
      }
    }
    __syncthreads();

    // ---- reload A with h_new (kt 0..7); a[8] still holds [inp_prev, 1] ---
#pragma unroll
    for (int kt = 0; kt < 8; ++kt) a[kt] = LOAD_A(kt);

    // ---- d = h_new . W_fc + b_fc (fc tile nt=64, column 0) ---------------
    v8f dacc = {};
#pragma unroll
    for (int kt = 0; kt < KTILES; ++kt) {
      v16h bfc = *(const v16h*)(bbase + (64 * KTILES + kt) * fragsz);
      dacc = __builtin_amdgcn_wmma_f32_16x16x32_f16(false, a[kt], false, bfc,
                                                    (short)0, dacc, false, false);
    }
    // column 0 lives in lanes 0 (rows 0-7) and 16 (rows 8-15)
    if (nlo == 0) {
#pragma unroll
      for (int v = 0; v < 8; ++v) {
        const int m = v + 8 * sel;
        float d = dacc[v];
        out[(size_t)(row_base + m) * SEQ + t] = d;
        hbuf[w][m][HID] = (_Float16)d;   // next step's input
      }
    }
    __syncthreads();
    a[8] = LOAD_A(8);                    // refresh [inp, 1] fragment
  }
#undef LOAD_A
}

// ---------------------------------------------------------------------------
extern "C" void kernel_launch(void* const* d_in, const int* in_sizes, int n_in,
                              void* d_out, int out_size, void* d_ws, size_t ws_size,
                              hipStream_t stream) {
  const float* x    = (const float*)d_in[0];
  const float* W_ih = (const float*)d_in[1];
  const float* W_hh = (const float*)d_in[2];
  const float* b_ih = (const float*)d_in[3];
  const float* b_hh = (const float*)d_in[4];
  const float* W_fc = (const float*)d_in[5];
  const float* b_fc = (const float*)d_in[6];
  _Float16* wpack = (_Float16*)d_ws;     // 65*9*512 f16 = 585 KiB
  float* out = (float*)d_out;

  const int packN = NTILES * KTILES * 512;
  hipLaunchKernelGGL(lstm_pack_weights, dim3((packN + 255) / 256), dim3(256), 0,
                     stream, W_ih, W_hh, b_ih, b_hh, W_fc, b_fc, wpack);

  const int blocks = BATCH / (16 * WAVES);   // 256 workgroups, 2 waves each
  hipLaunchKernelGGL(DepthMarkerPredictor_lstm_kernel, dim3(blocks),
                     dim3(32 * WAVES), 0, stream, x, wpack, out);
}
